// ConceptLayer_90683939488105
// MI455X (gfx1250) — matrix operand order repaired and checked by reading, not compiled, not run on hardware
//
#include <hip/hip_runtime.h>
#include <hip/hip_bf16.h>

// ---------------------------------------------------------------------------
// ConceptLayer fused implementation for gfx1250 (MI455X), bf16 WMMA + TDM.
//   K0a: W  -> bf16 (row k, col K=a*64+c)        [B-operand source layout]
//   K0b: x  -> bf16 transposed xbfT[b*64+c][j]   [TDM tile source for K1]
//   K1 : s_pre = T @ x  (causal Toeplitz GEMM, packed-pair LDS decay table,
//                        TDM double-buffered x tiles)
//   K2 : out = (x (x) s_pre) @ W^T + residual + LayerNorm, fused per 256
//        tokens; W chunks staged to LDS by TDM.
// ---------------------------------------------------------------------------

#define S_LEN 4096
#define D_DIM 64
#define LN_EPS 1e-3f

typedef __attribute__((ext_vector_type(16))) __bf16 v16bf;
typedef __attribute__((ext_vector_type(8)))  float  v8f;
typedef __attribute__((ext_vector_type(4)))  unsigned int u32x4;
typedef __attribute__((ext_vector_type(8)))  int          i32x8;
typedef __attribute__((ext_vector_type(4)))  int          i32x4;

union V16 {
    v16bf        v;
    unsigned int u[8];
    uint4        q[2];
};

__device__ __forceinline__ unsigned short f2bf(float f) {
    unsigned int u = __float_as_uint(f);
    unsigned int r = u + 0x7FFFu + ((u >> 16) & 1u);   // round-to-nearest-even
    return (unsigned short)(r >> 16);
}
// Hot-path pair pack: lower through HIP bf16 API so the backend can emit the
// hardware f32->bf16 pack-convert instead of ~6 bit-ops per dword.
__device__ __forceinline__ unsigned int pack2(float lo, float hi) {
    __hip_bfloat162 h = __float22bfloat162_rn(make_float2(lo, hi));
    union { __hip_bfloat162 h2; unsigned int u; } cvt;
    cvt.h2 = h;
    return cvt.u;
}

// Generic shared-memory pointer -> LDS byte offset (LDS aperture keeps the
// offset in addr[31:0] per CDNA5 flat-addressing rules).
__device__ __forceinline__ unsigned lds_off(const void* p) {
    return (unsigned)(unsigned long long)p;
}

// ---------------------------------------------------------------------------
// Tensor Data Mover: 2D bf16 tile load (global -> LDS), optional LDS padding.
// D# packing per CDNA5 ISA (groups 2/3 zeroed:
// tile_dim2=0 -> dims 2..4 unused). 6-arg builtin form (clang-23 lane).
// ---------------------------------------------------------------------------
__device__ __forceinline__ void tdm_load_2d_bf16(
    unsigned lds_byte_addr, const void* gaddr,
    unsigned tile_d0, unsigned tile_d1,
    unsigned tensor_d0, unsigned tensor_d1,
    unsigned long long stride0_elems,
    unsigned pad_en, unsigned pad_interval, unsigned pad_amount) {
    unsigned long long ga = (unsigned long long)gaddr;
    u32x4 g0;
    g0.x = 1u;                                   // count=1 (valid), user mode
    g0.y = lds_byte_addr;                        // [63:32] lds_addr
    g0.z = (unsigned)ga;                         // [95:64] global_addr lo
    g0.w = (unsigned)((ga >> 32) & 0x01FFFFFFull) | 0x80000000u;  // type=2
    i32x8 g1;
    g1[0] = (int)((1u << 16) |                   // data_size = 2 bytes
                  (pad_en << 20) | (pad_interval << 22) | (pad_amount << 25));
    g1[1] = (int)((tensor_d0 & 0xFFFFu) << 16);                 // dim0 lo16
    g1[2] = (int)((tensor_d0 >> 16) | ((tensor_d1 & 0xFFFFu) << 16));
    g1[3] = (int)((tensor_d1 >> 16) | (tile_d0 << 16));
    g1[4] = (int)(tile_d1 & 0xFFFFu);            // tile_dim2 = 0
    g1[5] = (int)(unsigned)(stride0_elems & 0xFFFFFFFFull);
    g1[6] = (int)(unsigned)((stride0_elems >> 32) & 0xFFFFull);  // stride1=0
    g1[7] = 0;
    i32x4 z4 = {0, 0, 0, 0};
    i32x8 z8 = {0, 0, 0, 0, 0, 0, 0, 0};
    __builtin_amdgcn_tensor_load_to_lds(g0, g1, z4, z4, z8, 0);
}

// ---------------------------------------------------------------------------
// Kernel 0a: concept_map (64 x 4096) -> bf16, row-major (WMMA B source).
// ---------------------------------------------------------------------------
__global__ __launch_bounds__(256) void cvt_w_kernel(const float* __restrict__ W,
                                                    unsigned short* __restrict__ Wb) {
    int idx = blockIdx.x * 256 + threadIdx.x;
    if (idx < D_DIM * D_DIM * D_DIM) Wb[idx] = f2bf(W[idx]);
}

// ---------------------------------------------------------------------------
// Kernel 0b: x (B,S,D) f32 -> xbfT[(b*64+c)*4096 + j] bf16 (transposed).
// One-shot 4 MB transpose; writes coalesced, reads L2-resident.
// ---------------------------------------------------------------------------
__global__ __launch_bounds__(256) void cvt_x_kernel(const float* __restrict__ x,
                                                    unsigned short* __restrict__ xbfT) {
    int idx = blockIdx.x * 256 + threadIdx.x;        // over B*D*S = 2M
    if (idx < 8 * D_DIM * S_LEN) {
        int j  = idx & (S_LEN - 1);
        int bc = idx >> 12;
        int c  = bc & 63;
        int b  = bc >> 6;
        xbfT[idx] = f2bf(x[((size_t)(b * S_LEN + j)) * D_DIM + c]);
    }
}

// ---------------------------------------------------------------------------
// Kernel 1: s_pre[b,i,c] = sum_{j<i} x[b,j,c] / (i-j)^2
// A (16x32 T-tile) from packed-pair bf16 decay tables (causal mask baked in);
// B (32x16 x-tile) TDM double-buffered into LDS with hardware row padding.
// ---------------------------------------------------------------------------
#define ROWS_WG   128
#define KSTEP     32
#define XT_STRIDE 40          // halves per column row (80B): 32 data + 8 TDM pad
#define TBL_N     2176        // packed dword entries per parity table

__global__ __launch_bounds__(256) void toeplitz_spre_kernel(
    const unsigned short* __restrict__ xbfT, float* __restrict__ spre) {
    __shared__ unsigned int   tpk[2][TBL_N];
    __shared__ unsigned short xT[2][D_DIM * XT_STRIDE];

    const int tid = threadIdx.x;
    const int wg  = blockIdx.x;
    const int b   = wg >> 5;                 // 32 row-blocks per batch
    const int rb  = wg & 31;
    const int i0  = rb * ROWS_WG;

    // Build packed decay tables (one-time per WG).
    for (int n = tid; n < TBL_N; n += 256) {
#pragma unroll
        for (int p = 0; p < 2; ++p) {
            int   d0 = 4096 + p - 2 * n;
            int   d1 = d0 - 1;
            float lo = (d0 > 0) ? 1.0f / ((float)d0 * (float)d0) : 0.0f;
            float hi = (d1 > 0) ? 1.0f / ((float)d1 * (float)d1) : 0.0f;
            tpk[p][n] = pack2(lo, hi);
        }
    }

    const int lane = tid & 31;
    const int w    = tid >> 5;               // wave id: M-tile
    const int m16  = lane & 15;
    const int hi16 = lane >> 4;
    const int i    = i0 + w * 16 + m16;      // this lane's T row
    const int g0   = hi16 * 8;               // K half-group base (A layout)

    const v8f zero8 = {0.f, 0.f, 0.f, 0.f, 0.f, 0.f, 0.f, 0.f};
    v8f acc[4] = {zero8, zero8, zero8, zero8};

    const int jend = i0 + ROWS_WG;
    const unsigned short* xb = xbfT + (size_t)b * D_DIM * S_LEN;

    // Prologue: TDM tile for j0 = 0 into buffer 0.
    if (w == 0)
        tdm_load_2d_bf16(lds_off(&xT[0][0]), xb, /*tile*/ KSTEP, D_DIM,
                         /*tensor*/ S_LEN, D_DIM, /*stride0*/ S_LEN,
                         /*pad: every 64B (+16B) -> 40-half rows*/ 1, 3, 3);

    int cur = 0;
    for (int j0 = 0; j0 < jend; j0 += KSTEP) {
        if (w == 0) __builtin_amdgcn_s_wait_tensorcnt(0);
        __syncthreads();   // xT[cur] ready; prior reads of xT[cur^1] retired
        if (w == 0 && (j0 + KSTEP) < jend)
            tdm_load_2d_bf16(lds_off(&xT[cur ^ 1][0]), xb + (j0 + KSTEP),
                             KSTEP, D_DIM, (unsigned)(S_LEN - (j0 + KSTEP)),
                             D_DIM, S_LEN, 1, 3, 3);

        // A operand: Toeplitz tile via packed table, 8 ds dwords per lane.
        const int p     = (i - j0) & 1;
        const int nbase = (4096 + p - i + j0) >> 1;
        V16 A;
#pragma unroll
        for (int q = 0; q < 8; ++q) {
            int K0 = g0 + (q & 3) * 2 + (q >> 2) * 16;
            A.u[q] = tpk[p][nbase + (K0 >> 1)];
        }

        // 4 N-tiles of x columns; A reused across all 4 WMMAs.
#pragma unroll
        for (int nt = 0; nt < 4; ++nt) {
            V16 Bv;
            const uint4* bq =
                (const uint4*)&xT[cur][(nt * 16 + m16) * XT_STRIDE + hi16 * 8];
            Bv.q[0] = bq[0];
            Bv.q[1] = bq[2];
            acc[nt] = __builtin_amdgcn_wmma_f32_16x16x32_bf16(
                false, A.v, false, Bv.v, (short)0, acc[nt], false, false);
        }
        cur ^= 1;
    }

    // Epilogue: scatter f32 accumulators to spre (coalesced across lanes in N).
    const int orow0 = b * S_LEN + i0 + w * 16;
#pragma unroll
    for (int nt = 0; nt < 4; ++nt) {
#pragma unroll
        for (int v = 0; v < 8; ++v) {
            int rr = v + 8 * hi16;
            spre[(size_t)(orow0 + rr) * D_DIM + nt * 16 + m16] = acc[nt][v];
        }
    }
}

// ---------------------------------------------------------------------------
// Kernel 2: fused  out = (x (x) s_pre) @ W^T ; r = out + x ; LayerNorm(r).
// GEMM view: M = 256 tokens/WG, N = 64, K = 4096 in 32 chunks of 128.
// outer chunk (256x128 bf16) built in LDS; W chunk (64x128 bf16) TDM-staged
// into LDS (hardware padding -> 136-half rows, bank-conflict free);
// 16x16 f32 accumulators persist in VGPRs across all chunks.
// ---------------------------------------------------------------------------
#define TOK_WG     256
#define CK         128                    // K-chunk (halves)
#define NCHUNK     32
#define OUT_STRIDE 136                    // halves (272B, 16B-aligned)
#define R_STRIDE   68                     // f32 (272B, 16B-aligned, bank-padded)

#define SM_XL   0                                  // float xl[256*64]   : 64 KB
#define SM_SL   65536                              // float sl[256*64]   : 64 KB
#define SM_OUT  131072                             // bf16 outer[256*136]: 68 KB
#define SM_WL   (131072 + TOK_WG * OUT_STRIDE * 2) // bf16 Wl[64*136]    : 17 KB
#define SM_R    65536                              // float rt[256*68] (alias)
#define SM_SIZE (SM_WL + D_DIM * OUT_STRIDE * 2)

__global__ __launch_bounds__(256) void bilinear_ln_kernel(
    const float* __restrict__ x, const float* __restrict__ spre,
    const unsigned short* __restrict__ Wb, const float* __restrict__ gamma,
    const float* __restrict__ beta, float* __restrict__ out) {
    extern __shared__ unsigned char smem[];
    float*          xl    = (float*)(smem + SM_XL);
    float*          sl    = (float*)(smem + SM_SL);
    unsigned short* outer = (unsigned short*)(smem + SM_OUT);
    unsigned short* Wl    = (unsigned short*)(smem + SM_WL);
    float*          rt    = (float*)(smem + SM_R);

    const int    tid = threadIdx.x;
    const size_t t0  = (size_t)blockIdx.x * TOK_WG;

    // Stage x and s_pre tiles (f32).
    for (int e = tid; e < TOK_WG * D_DIM; e += 256) {
        xl[e] = x[t0 * D_DIM + e];
        sl[e] = spre[t0 * D_DIM + e];
    }

    const int lane = tid & 31;
    const int w    = tid >> 5;
    const int ng   = w & 1;                 // N-group: cols ng*32 .. +31
    const int mg   = w >> 1;                // M-group: tokens mg*64 .. +63
    const int m16  = lane & 15;
    const int hi16 = lane >> 4;

    const v8f zero8 = {0.f, 0.f, 0.f, 0.f, 0.f, 0.f, 0.f, 0.f};
    v8f acc[4][2] = {{zero8, zero8}, {zero8, zero8}, {zero8, zero8}, {zero8, zero8}};

    for (int kc = 0; kc < NCHUNK; ++kc) {
        __syncthreads();   // prior readers of outer/Wl done; xl/sl staged
        // TDM: stage this chunk's W stripe (64 rows x 128 halves) into LDS,
        // padded every 256B by 16B -> 136-half rows. Overlaps outer build.
        if (w == 0)
            tdm_load_2d_bf16(lds_off(Wl), Wb + (size_t)kc * CK,
                             /*tile*/ CK, D_DIM,
                             /*tensor*/ (unsigned)(D_DIM * D_DIM - kc * CK),
                             D_DIM, /*stride0*/ D_DIM * D_DIM,
                             /*pad: every 256B (+16B)*/ 1, 5, 3);

        // Build outer chunk: thread == token row; a = kc*2 {+0,+1}, c = 0..63.
        {
            const int mrow = tid;
            float xa0 = xl[mrow * D_DIM + kc * 2 + 0];
            float xa1 = xl[mrow * D_DIM + kc * 2 + 1];
            unsigned short* orow = outer + mrow * OUT_STRIDE;
#pragma unroll
            for (int c = 0; c < D_DIM; c += 2) {
                float s0 = sl[mrow * D_DIM + c];
                float s1 = sl[mrow * D_DIM + c + 1];
                *(unsigned int*)(orow + c)      = pack2(xa0 * s0, xa0 * s1);
                *(unsigned int*)(orow + 64 + c) = pack2(xa1 * s0, xa1 * s1);
            }
        }
        if (w == 0) __builtin_amdgcn_s_wait_tensorcnt(0);
        __syncthreads();   // outer + Wl ready

#pragma unroll
        for (int ks = 0; ks < 4; ++ks) {
            const int kk = ks * 32;
            // B operands from TDM-staged W (2x ds_load_b128 per lane).
            V16 Bt[2];
#pragma unroll
            for (int nt = 0; nt < 2; ++nt) {
                int n = ng * 32 + nt * 16 + m16;
                const uint4* bq =
                    (const uint4*)(Wl + n * OUT_STRIDE + kk + hi16 * 8);
                Bt[nt].q[0] = bq[0];
                Bt[nt].q[1] = bq[2];
            }
#pragma unroll
            for (int mt = 0; mt < 4; ++mt) {
                int mrow = mg * 64 + mt * 16 + m16;
                const unsigned short* ap = outer + mrow * OUT_STRIDE + kk + hi16 * 8;
                V16 A;
                A.q[0] = *(const uint4*)ap;
                A.q[1] = *(const uint4*)(ap + 16);
#pragma unroll
                for (int nt = 0; nt < 2; ++nt)
                    acc[mt][nt] = __builtin_amdgcn_wmma_f32_16x16x32_bf16(
                        false, A.v, false, Bt[nt].v, (short)0, acc[mt][nt],
                        false, false);
            }
        }
    }
    __syncthreads();

    // r = out + x  -> bank-padded LDS tile (rt aliases sl/outer; xl stays live).
#pragma unroll
    for (int mt = 0; mt < 4; ++mt)
#pragma unroll
        for (int nt = 0; nt < 2; ++nt)
#pragma unroll
            for (int v = 0; v < 8; ++v) {
                int m = mg * 64 + mt * 16 + v + 8 * hi16;
                int c = ng * 32 + nt * 16 + m16;
                rt[m * R_STRIDE + c] = acc[mt][nt][v] + xl[m * D_DIM + c];
            }
    __syncthreads();

    // LayerNorm: one token per thread.
    {
        const int    m  = tid;
        const float* rr = rt + m * R_STRIDE;
        float sum = 0.f, sq = 0.f;
#pragma unroll
        for (int c = 0; c < D_DIM; c += 4) {
            float4 v = *(const float4*)(rr + c);
            sum += v.x + v.y + v.z + v.w;
            sq  += v.x * v.x + v.y * v.y + v.z * v.z + v.w * v.w;
        }
        float mu  = sum * (1.0f / D_DIM);
        float var = sq * (1.0f / D_DIM) - mu * mu;
        float inv = rsqrtf(var + LN_EPS);
        float* orow = out + (t0 + m) * D_DIM;
#pragma unroll
        for (int c = 0; c < D_DIM; c += 4) {
            float4 v = *(const float4*)(rr + c);
            float4 y;
            y.x = (v.x - mu) * inv * gamma[c + 0] + beta[c + 0];
            y.y = (v.y - mu) * inv * gamma[c + 1] + beta[c + 1];
            y.z = (v.z - mu) * inv * gamma[c + 2] + beta[c + 2];
            y.w = (v.w - mu) * inv * gamma[c + 3] + beta[c + 3];
            *(float4*)(orow + c) = y;
        }
    }
}

// ---------------------------------------------------------------------------
// Launch
// ---------------------------------------------------------------------------
extern "C" void kernel_launch(void* const* d_in, const int* in_sizes, int n_in,
                              void* d_out, int out_size, void* d_ws, size_t ws_size,
                              hipStream_t stream) {
    (void)in_sizes; (void)n_in; (void)out_size; (void)ws_size;
    const float* x     = (const float*)d_in[0];
    const float* W     = (const float*)d_in[1];
    const float* gamma = (const float*)d_in[2];
    const float* beta  = (const float*)d_in[3];
    float*       out   = (float*)d_out;

    const size_t wb_bytes  = (size_t)D_DIM * D_DIM * D_DIM * 2;   // 512 KB
    const size_t xbf_bytes = (size_t)8 * D_DIM * S_LEN * 2;       // 4 MB
    unsigned short* Wb   = (unsigned short*)d_ws;
    unsigned short* xbfT = (unsigned short*)((char*)d_ws + wb_bytes);
    float*          spre = (float*)((char*)d_ws + wb_bytes + xbf_bytes);  // 8 MB

    cvt_w_kernel<<<(D_DIM * D_DIM * D_DIM + 255) / 256, 256, 0, stream>>>(W, Wb);
    cvt_x_kernel<<<(8 * D_DIM * S_LEN + 255) / 256, 256, 0, stream>>>(x, xbfT);
    toeplitz_spre_kernel<<<8 * (S_LEN / ROWS_WG), 256, 0, stream>>>(xbfT, spre);

    (void)hipFuncSetAttribute(reinterpret_cast<const void*>(bilinear_ln_kernel),
                              hipFuncAttributeMaxDynamicSharedMemorySize,
                              (int)SM_SIZE);
    bilinear_ln_kernel<<<(8 * S_LEN) / TOK_WG, 256, SM_SIZE, stream>>>(
        x, spre, Wb, gamma, beta, out);
}
